// GEWA_34823594836440
// MI455X (gfx1250) — matrix operand order repaired and checked by reading, not compiled, not run on hardware
//
#include <hip/hip_runtime.h>
#include <cstdint>
#include <cstddef>

#define GN_N    50000
#define GN_E    800000
#define GN_FIN  1433
#define GN_H    8
#define GN_C    7
#define GN_PH   16
#define GN_L    3
#define GN_D0   56      // H*C
#define GN_DH   128     // PH*H

typedef __attribute__((ext_vector_type(16))) _Float16 v16h;
typedef __attribute__((ext_vector_type(8)))  _Float16 v8h;
typedef __attribute__((ext_vector_type(8)))  float    v8f;
typedef __attribute__((ext_vector_type(4)))  float    v4f;

// =====================================================================
// Weight pre-pack: W[K,Nout] f32 row-major -> WT[Npad,Kpad] f16
// (transposed + zero padded so GEMM B loads are unconditional 16B vectors)
// =====================================================================
__global__ void gnn_wpack(const float* __restrict__ W, _Float16* __restrict__ WT,
                          int K, int Nout, int Kpad, int total) {
  int i = blockIdx.x * blockDim.x + threadIdx.x;
  if (i >= total) return;
  int n = i / Kpad, k = i - n * Kpad;
  float v = (n < Nout && k < K) ? W[(size_t)k * Nout + n] : 0.0f;
  WT[i] = (_Float16)v;
}

// =====================================================================
// WMMA GEMM: C[M,Nout] = A[M,K] @ W[K,Nout] (+bias)(+relu)
// One wave computes a 16-row stripe x NT 16-col tiles (v_wmma_f32_16x16x32_f16,
// NT accumulators reuse each A fragment). Contract: M % 16 == 0.
// Per k-step: ALL NT B fragments are preloaded (one clause, one wait), then
// the NT WMMAs issue back-to-back.
// Fragment layouts per ISA 7.12.2:
//   A 16x32: lanes 0-15 hold K {0..7,16..23}, lanes 16-31 {8..15,24..31}
//   B 32x16: lanes 0-15 hold K {0..7,16..23}, lanes 16-31 {8..15,24..31}
//   C 16x16: lane%16 = N, vgpr r = M=r (+8 for lanes 16-31)
// =====================================================================
template <int NT, bool VEC4>
__global__ void gnn_wmma_gemm(const float* __restrict__ A,
                              const _Float16* __restrict__ WT,
                              const float* __restrict__ bias,
                              float* __restrict__ C,
                              int M, int K, int Kpad, int Nout,
                              int relu, int addb) {
  const int lane  = threadIdx.x & 31;
  const int wave  = threadIdx.x >> 5;
  const int mtile = blockIdx.x * 8 + wave;
  if (mtile * 16 >= M) return;            // wave-uniform exit
  const int hi  = lane >> 4;
  const int l16 = lane & 15;
  const float* Arow = A + (size_t)(mtile * 16 + l16) * K;
  const _Float16* Wlane = WT + (size_t)l16 * Kpad + hi * 8;

  v8f acc[NT];
#pragma unroll
  for (int t = 0; t < NT; ++t) acc[t] = (v8f){};

  const int Kmain = K & ~31;
  int kk = 0;
  for (; kk < Kmain; kk += 32) {
    // ---- issue all B loads for this k-step first (batched) ----
    v8h b0[NT], b1[NT];
#pragma unroll
    for (int t = 0; t < NT; ++t) {
      const v8h* bp = (const v8h*)(Wlane + (size_t)t * 16 * Kpad + kk);
      b0[t] = bp[0];
      b1[t] = bp[2];
    }
    // ---- A fragment ----
    v16h a;
    if constexpr (VEC4) {                 // rows 16B aligned (K%4==0)
      const v4f* p0 = (const v4f*)(Arow + kk + hi * 8);
      const v4f* p1 = (const v4f*)(Arow + kk + 16 + hi * 8);
      v4f r0 = p0[0], r1 = p0[1], r2 = p1[0], r3 = p1[1];
#pragma unroll
      for (int j = 0; j < 4; ++j) {
        a[j]      = (_Float16)r0[j];
        a[4 + j]  = (_Float16)r1[j];
        a[8 + j]  = (_Float16)r2[j];
        a[12 + j] = (_Float16)r3[j];
      }
    } else {                              // unconditional scalar loads
#pragma unroll
      for (int j = 0; j < 16; ++j) {
        int ka = kk + ((j < 8) ? (hi * 8 + j) : (8 + hi * 8 + j));
        a[j] = (_Float16)Arow[ka];
      }
    }
    // ---- WMMA chain ----
#pragma unroll
    for (int t = 0; t < NT; ++t) {
      v16h b;
#pragma unroll
      for (int j = 0; j < 8; ++j) { b[j] = b0[t][j]; b[8 + j] = b1[t][j]; }
      acc[t] = __builtin_amdgcn_wmma_f32_16x16x32_f16(false, a, false, b,
                                                      (short)0, acc[t], false, false);
    }
  }
  if (kk < K) {                           // K tail: clamped address + select
    v8h b0[NT], b1[NT];
#pragma unroll
    for (int t = 0; t < NT; ++t) {
      const v8h* bp = (const v8h*)(Wlane + (size_t)t * 16 * Kpad + kk);
      b0[t] = bp[0];
      b1[t] = bp[2];
    }
    v16h a;
#pragma unroll
    for (int j = 0; j < 16; ++j) {
      int ka = kk + ((j < 8) ? (hi * 8 + j) : (8 + hi * 8 + j));
      float va = Arow[ka < K ? ka : 0];   // always-valid load, no exec churn
      a[j] = (_Float16)(ka < K ? va : 0.0f);
    }
#pragma unroll
    for (int t = 0; t < NT; ++t) {
      v16h b;
#pragma unroll
      for (int j = 0; j < 8; ++j) { b[j] = b0[t][j]; b[8 + j] = b1[t][j]; }
      acc[t] = __builtin_amdgcn_wmma_f32_16x16x32_f16(false, a, false, b,
                                                      (short)0, acc[t], false, false);
    }
  }
  // epilogue (M%16==0 -> only column guard needed)
#pragma unroll
  for (int t = 0; t < NT; ++t) {
    int colB = t * 16 + l16;
    if (colB < Nout) {
      float bv = addb ? bias[colB] : 0.0f;
#pragma unroll
      for (int r = 0; r < 8; ++r) {
        int m = mtile * 16 + r + hi * 8;
        float v = acc[t][r] + bv;
        if (relu) v = fmaxf(v, 0.0f);
        C[(size_t)m * Nout + colB] = v;
      }
    }
  }
}

// ============================ utility ================================
__global__ void gnn_fill(float* p, float v, int n) {
  int i = blockIdx.x * blockDim.x + threadIdx.x;
  if (i < n) p[i] = v;
}

__device__ __forceinline__ void gnn_edge(const int* __restrict__ ei, int e,
                                         int& s, int& d) {
  if (e < GN_E) { s = ei[e]; d = ei[GN_E + e]; }
  else          { s = d = e - GN_E; }             // implicit self-loop
}

__device__ __forceinline__ float gnn_lrelu(float x) {
  return x > 0.0f ? x : 0.2f * x;
}

__device__ __forceinline__ void gnn_atomic_maxf(float* addr, float v) {
  // exact float max via int max (v>=0) / uint min (v<0); init must be -inf
  if (v >= 0.0f) atomicMax((int*)addr, __float_as_int(v));
  else           atomicMin((unsigned int*)addr, __float_as_uint(v));
}

// ======================= degree / normalization ======================
__global__ void gnn_deg(const int* __restrict__ ei, float* deg, int total) {
  int i = blockIdx.x * blockDim.x + threadIdx.x;
  if (i >= total) return;
  int s, d; gnn_edge(ei, i, s, d);
  atomicAdd(&deg[d], 1.0f);
}

__global__ void gnn_dinv(float* deg, int n) {
  int i = blockIdx.x * blockDim.x + threadIdx.x;
  if (i < n) deg[i] = rsqrtf(fmaxf(deg[i], 1.0f));
}

// ============================ GCN propagate ==========================
__global__ void gnn_gcn_prop(const int* __restrict__ ei,
                             const float* __restrict__ dinv,
                             const float* __restrict__ t,
                             float* __restrict__ agg,
                             int dh, int nchunk, int total) {
  int tid = blockIdx.x * blockDim.x + threadIdx.x;
  if (tid >= total) return;
  int e = tid / nchunk, ch = tid - e * nchunk;
  int s, d; gnn_edge(ei, e, s, d);
  float nm = dinv[s] * dinv[d];
  int c0 = ch * 8, c1 = (c0 + 8 < dh) ? c0 + 8 : dh;
  const float* ts = t + (size_t)s * dh;
  float* ad = agg + (size_t)d * dh;
  for (int c = c0; c < c1; ++c) atomicAdd(&ad[c], ts[c] * nm);
}

__global__ void gnn_bias_act(const float* __restrict__ agg,
                             const float* __restrict__ b,
                             float* __restrict__ h, int dh, int relu, int total) {
  int i = blockIdx.x * blockDim.x + threadIdx.x;
  if (i >= total) return;
  float v = agg[i] + b[i % dh];
  if (relu) v = fmaxf(v, 0.0f);
  h[i] = v;
}

// ================================ GAT ================================
__global__ void gnn_attn_coef(const float* __restrict__ hh,
                              const float* __restrict__ as,
                              const float* __restrict__ ad,
                              float* __restrict__ es, float* __restrict__ ed,
                              int total /* N*H */) {
  int i = blockIdx.x * blockDim.x + threadIdx.x;
  if (i >= total) return;
  int hd = i & (GN_H - 1), n = i >> 3;
  const float* p = hh + (size_t)n * GN_DH + hd * GN_PH;
  float s = 0.0f, d = 0.0f;
#pragma unroll
  for (int c = 0; c < GN_PH; ++c) {
    s += p[c] * as[hd * GN_PH + c];
    d += p[c] * ad[hd * GN_PH + c];
  }
  es[i] = s; ed[i] = d;
}

__global__ void gnn_attn_init(float* emax, float* denom, int total) {
  int i = blockIdx.x * blockDim.x + threadIdx.x;
  if (i >= total) return;
  emax[i] = -3.402823466e38f;
  denom[i] = 0.0f;
}

__global__ void gnn_gat_max(const int* __restrict__ ei,
                            const float* __restrict__ es,
                            const float* __restrict__ ed,
                            float* __restrict__ emax, int total) {
  int tid = blockIdx.x * blockDim.x + threadIdx.x;
  if (tid >= total) return;
  int e = tid >> 3, hd = tid & 7;
  int s, d; gnn_edge(ei, e, s, d);
  float v = gnn_lrelu(es[s * GN_H + hd] + ed[d * GN_H + hd]);
  gnn_atomic_maxf(&emax[d * GN_H + hd], v);
}

__global__ void gnn_gat_denom(const int* __restrict__ ei,
                              const float* __restrict__ es,
                              const float* __restrict__ ed,
                              const float* __restrict__ emax,
                              float* __restrict__ denom, int total) {
  int tid = blockIdx.x * blockDim.x + threadIdx.x;
  if (tid >= total) return;
  int e = tid >> 3, hd = tid & 7;
  int s, d; gnn_edge(ei, e, s, d);
  float v = gnn_lrelu(es[s * GN_H + hd] + ed[d * GN_H + hd]);
  atomicAdd(&denom[d * GN_H + hd], expf(v - emax[d * GN_H + hd]));
}

__global__ void gnn_gat_acc(const int* __restrict__ ei,
                            const float* __restrict__ es,
                            const float* __restrict__ ed,
                            const float* __restrict__ emax,
                            const float* __restrict__ denom,
                            const float* __restrict__ hh,
                            float* __restrict__ agg, int total) {
  int tid = blockIdx.x * blockDim.x + threadIdx.x;
  if (tid >= total) return;
  int e = tid >> 3, hd = tid & 7;
  int s, d; gnn_edge(ei, e, s, d);
  float v = gnn_lrelu(es[s * GN_H + hd] + ed[d * GN_H + hd]);
  float alpha = expf(v - emax[d * GN_H + hd]) / denom[d * GN_H + hd];
  const float* src = hh + (size_t)s * GN_DH + hd * GN_PH;
  float* dst = agg + (size_t)d * GN_DH + hd * GN_PH;
#pragma unroll
  for (int c = 0; c < GN_PH; ++c) atomicAdd(&dst[c], src[c] * alpha);
}

__global__ void gnn_gat_final(const float* __restrict__ agg,
                              const float* __restrict__ gb,
                              float* __restrict__ h, int total /* N*PH */) {
  int i = blockIdx.x * blockDim.x + threadIdx.x;
  if (i >= total) return;
  int n = i >> 4, c = i & (GN_PH - 1);
  float s = 0.0f;
#pragma unroll
  for (int hd = 0; hd < GN_H; ++hd) s += agg[(size_t)n * GN_DH + hd * GN_PH + c];
  h[i] = s * (1.0f / GN_H) + gb[c];
}

// =============================== driver ==============================
static inline int gnn_cdiv(int a, int b) { return (a + b - 1) / b; }

extern "C" void kernel_launch(void* const* d_in, const int* in_sizes, int n_in,
                              void* d_out, int out_size, void* d_ws, size_t ws_size,
                              hipStream_t stream) {
  (void)in_sizes; (void)n_in; (void)out_size; (void)ws_size;
  const float* x      = (const float*)d_in[0];
  const int*   ei     = (const int*)  d_in[1];
  const float* w_feat = (const float*)d_in[2];
  const float* b_feat = (const float*)d_in[3];
  const float* gcn0_w = (const float*)d_in[4];
  const float* gcn0_b = (const float*)d_in[5];
  const float* gcn_w  = (const float*)d_in[6];
  const float* gcn_b  = (const float*)d_in[7];
  const float* gat_w  = (const float*)d_in[8];
  const float* gat_as = (const float*)d_in[9];
  const float* gat_ad = (const float*)d_in[10];
  const float* gat_b  = (const float*)d_in[11];
  const float* mas_w  = (const float*)d_in[12];
  const float* mas_b  = (const float*)d_in[13];
  const float* out_w  = (const float*)d_in[14];
  const float* out_b  = (const float*)d_in[15];
  float* out = (float*)d_out;

  // 256B-aligned workspace carving
  char* wsb = (char*)d_ws;
  auto walloc = [&](size_t nbytes) {
    char* p = wsb; wsb += (nbytes + 255) & ~(size_t)255; return p;
  };
  float*    dinv = (float*)walloc((size_t)GN_N * 4);
  float*    hbuf = (float*)walloc((size_t)GN_N * GN_DH * 4);
  float*    tbuf = (float*)walloc((size_t)GN_N * GN_DH * 4);
  float*    agg  = (float*)walloc((size_t)GN_N * GN_DH * 4);
  float*    es   = (float*)walloc((size_t)GN_N * GN_H * 4);
  float*    ed   = (float*)walloc((size_t)GN_N * GN_H * 4);
  float*    emax = (float*)walloc((size_t)GN_N * GN_H * 4);
  float*    deno = (float*)walloc((size_t)GN_N * GN_H * 4);
  _Float16* wT   = (_Float16*)walloc((size_t)64 * 1440 * 2);  // max Npad*Kpad

  const int TOTE = GN_E + GN_N;
  const int BLK = 256;

  // degrees -> D^-1/2
  gnn_fill<<<gnn_cdiv(GN_N, BLK), BLK, 0, stream>>>(dinv, 0.0f, GN_N);
  gnn_deg<<<gnn_cdiv(TOTE, BLK), BLK, 0, stream>>>(ei, dinv, TOTE);
  gnn_dinv<<<gnn_cdiv(GN_N, BLK), BLK, 0, stream>>>(dinv, GN_N);

  auto gemm = [&](const float* A, const float* W, const float* bias, float* C,
                  int M, int K, int Nout, int relu, int addb) {
    int Kpad = (K + 31) & ~31;
    int Npad = (Nout + 15) & ~15;
    int NT   = Npad / 16;
    int ptot = Npad * Kpad;
    gnn_wpack<<<gnn_cdiv(ptot, BLK), BLK, 0, stream>>>(W, wT, K, Nout, Kpad, ptot);
    dim3 g(gnn_cdiv(M / 16, 8));
    bool v4 = (K % 4) == 0;
    if      (NT == 8 &&  v4) gnn_wmma_gemm<8, true ><<<g, BLK, 0, stream>>>(A, wT, bias, C, M, K, Kpad, Nout, relu, addb);
    else if (NT == 8 && !v4) gnn_wmma_gemm<8, false><<<g, BLK, 0, stream>>>(A, wT, bias, C, M, K, Kpad, Nout, relu, addb);
    else if (NT == 4 &&  v4) gnn_wmma_gemm<4, true ><<<g, BLK, 0, stream>>>(A, wT, bias, C, M, K, Kpad, Nout, relu, addb);
    else if (NT == 4 && !v4) gnn_wmma_gemm<4, false><<<g, BLK, 0, stream>>>(A, wT, bias, C, M, K, Kpad, Nout, relu, addb);
    else if (v4)             gnn_wmma_gemm<1, true ><<<g, BLK, 0, stream>>>(A, wT, bias, C, M, K, Kpad, Nout, relu, addb);
    else                     gnn_wmma_gemm<1, false><<<g, BLK, 0, stream>>>(A, wT, bias, C, M, K, Kpad, Nout, relu, addb);
  };

  auto run_gcn = [&](const float* hin, int kin, const float* W, const float* b,
                     int dout, float* hout) {
    gemm(hin, W, nullptr, tbuf, GN_N, kin, dout, 0, 0);
    gnn_fill<<<gnn_cdiv(GN_N * dout, BLK), BLK, 0, stream>>>(agg, 0.0f, GN_N * dout);
    int nchunk = (dout + 7) / 8;
    int tot = TOTE * nchunk;
    gnn_gcn_prop<<<gnn_cdiv(tot, BLK), BLK, 0, stream>>>(ei, dinv, tbuf, agg,
                                                         dout, nchunk, tot);
    gnn_bias_act<<<gnn_cdiv(GN_N * dout, BLK), BLK, 0, stream>>>(agg, b, hout,
                                                                 dout, 1, GN_N * dout);
  };

  auto run_gat = [&](int m, int i) {
    size_t li = (size_t)(m * GN_L + i);
    gemm(hbuf, gat_w + li * GN_DH * GN_DH, nullptr, tbuf, GN_N, GN_DH, GN_DH, 0, 0);
    gnn_attn_coef<<<gnn_cdiv(GN_N * GN_H, BLK), BLK, 0, stream>>>(
        tbuf, gat_as + li * GN_H * GN_PH, gat_ad + li * GN_H * GN_PH,
        es, ed, GN_N * GN_H);
    gnn_attn_init<<<gnn_cdiv(GN_N * GN_H, BLK), BLK, 0, stream>>>(emax, deno,
                                                                  GN_N * GN_H);
    int tot = TOTE * GN_H;
    gnn_gat_max<<<gnn_cdiv(tot, BLK), BLK, 0, stream>>>(ei, es, ed, emax, tot);
    gnn_gat_denom<<<gnn_cdiv(tot, BLK), BLK, 0, stream>>>(ei, es, ed, emax, deno, tot);
    gnn_fill<<<gnn_cdiv(GN_N * GN_DH, BLK), BLK, 0, stream>>>(agg, 0.0f, GN_N * GN_DH);
    gnn_gat_acc<<<gnn_cdiv(tot, BLK), BLK, 0, stream>>>(ei, es, ed, emax, deno,
                                                        tbuf, agg, tot);
    gnn_gat_final<<<gnn_cdiv(GN_N * GN_PH, BLK), BLK, 0, stream>>>(
        agg, gat_b + li * GN_PH, hbuf, GN_N * GN_PH);
  };

  // GNFE: GCN(1433 -> 56) + relu
  run_gcn(x, GN_FIN, w_feat, b_feat, GN_D0, hbuf);

  for (int m = 0; m < GN_L; ++m) {
    for (int i = 0; i < GN_L; ++i) {
      if (i == 0)
        run_gcn(hbuf, GN_D0, gcn0_w + (size_t)m * GN_D0 * GN_DH,
                gcn0_b + (size_t)m * GN_DH, GN_DH, hbuf);
      else
        run_gcn(hbuf, GN_PH,
                gcn_w + (size_t)(m * (GN_L - 1) + i - 1) * GN_PH * GN_DH,
                gcn_b + (size_t)(m * (GN_L - 1) + i - 1) * GN_DH, GN_DH, hbuf);
      run_gat(m, i);
    }
    // MAS projection: [N,16] @ [16,56] + bias -> swap ping-pong buffers
    gemm(hbuf, mas_w + (size_t)m * GN_PH * GN_D0, mas_b + (size_t)m * GN_D0,
         tbuf, GN_N, GN_PH, GN_D0, 0, 1);
    float* tmp = hbuf; hbuf = tbuf; tbuf = tmp;
  }

  // classifier: [N,56] @ [56,7] + bias -> d_out
  gemm(hbuf, out_w, out_b, out, GN_N, GN_D0, GN_C, 0, 1);
}
